// Sg2ImModel_3530463117739
// MI455X (gfx1250) — matrix operand order repaired and checked by reading, not compiled, not run on hardware
//
#include <hip/hip_runtime.h>
#include <stdint.h>

// ---------------- bf16 helpers (bit-level, RNE) ----------------
typedef __attribute__((ext_vector_type(16))) __bf16 v16bf;
typedef __attribute__((ext_vector_type(8)))  float  v8f;

union Frag { v16bf v; uint4 q[2]; };

__device__ __forceinline__ unsigned short f2bf(float f) {
  unsigned int u = __float_as_uint(f);
  u += 0x7FFFu + ((u >> 16) & 1u);          // round-to-nearest-even
  return (unsigned short)(u >> 16);
}
__device__ __forceinline__ float bf2f(unsigned short h) {
  return __uint_as_float(((unsigned int)h) << 16);
}

// ---------------- WMMA GEMM tile config ----------------
// C(M,N) = A(M,K)[bf16,row-major] x B(K,N) with B pre-transposed Bt[N][K].
// Block: 256 threads = 8 waves arranged 4(M) x 2(N).
// Tile: 128(M) x 128(N), K-step 32.
// Each wave: 2 A-frags x 4 B-frags -> 8 v_wmma per K-step (12 ds_load_b128).
#define TM 128
#define TN 128
#define TK 32

__device__ __forceinline__ void wmma_mainloop(
    const unsigned short* __restrict__ A,
    const unsigned short* __restrict__ Bt,
    int M, int K, int tileM, int tileN,
    unsigned short* As, unsigned short* Bs,
    v8f acc[2][4])
{
  const int tidx  = threadIdx.x;
  const int wave  = tidx >> 5;
  const int lane  = tidx & 31;
  const int l15   = lane & 15;
  const int halfl = lane >> 4;
  const int mw    = wave >> 1;   // 0..3 : 32-row slab
  const int nw    = wave & 1;    // 0..1 : 64-col slab

  const int row  = tidx >> 1;    // tile stage: 16 elems per thread
  const int half = tidx & 1;

  for (int k0 = 0; k0 < K; k0 += TK) {
    __syncthreads();
    { // A tile 128x32 (zero-fill past M)
      const int grow = tileM + row;
      uint4 z0 = make_uint4(0u,0u,0u,0u), z1 = z0;
      if (grow < M) {
        const uint4* src = (const uint4*)(A + (size_t)grow * K + k0 + half * 16);
        z0 = src[0]; z1 = src[1];
        if (k0 + TK < K)
          __builtin_prefetch(A + (size_t)grow * K + k0 + TK, 0, 0); // global_prefetch
      }
      uint4* dst = (uint4*)(As + row * TK + half * 16);
      dst[0] = z0; dst[1] = z1;
    }
    { // B tile: Bs[n][k], n in [0,128) — N always a multiple of 128 here
      const uint4* src = (const uint4*)(Bt + (size_t)(tileN + row) * K + k0 + half * 16);
      uint4 b0 = src[0], b1 = src[1];
      uint4* dst = (uint4*)(Bs + row * TK + half * 16);
      dst[0] = b0; dst[1] = b1;
    }
    __syncthreads();

    // Load ALL fragments first so ds_loads clause up and WMMAs issue back-to-back.
    // A frag (16-bit A 16x32): lanes 0-15: M=lane, K={0..7,16..23};
    //                          lanes 16-31: M=lane-16, K={8..15,24..31}
    Frag fa[2];
#pragma unroll
    for (int mi = 0; mi < 2; ++mi) {
      const unsigned short* arow = As + (mw * 32 + mi * 16 + l15) * TK + halfl * 8;
      fa[mi].q[0] = *(const uint4*)(arow);
      fa[mi].q[1] = *(const uint4*)(arow + 16);
    }
    // B frag (16-bit B 32x16): lanes 0-15: N=lane, K={0..7,16..23};
    //                          lanes 16-31: N=lane-16, K={8..15,24..31}
    Frag fb[4];
#pragma unroll
    for (int nt = 0; nt < 4; ++nt) {
      const unsigned short* bcol = Bs + (nw * 64 + nt * 16 + l15) * TK + halfl * 8;
      fb[nt].q[0] = *(const uint4*)(bcol);
      fb[nt].q[1] = *(const uint4*)(bcol + 16);
    }
#pragma unroll
    for (int mi = 0; mi < 2; ++mi)
#pragma unroll
      for (int nt = 0; nt < 4; ++nt)
        acc[mi][nt] = __builtin_amdgcn_wmma_f32_16x16x32_bf16(
            false, fa[mi].v, false, fb[nt].v, (short)0, acc[mi][nt], false, false);
  }
}

// GEMM + bias (+ optional ReLU), bf16 output
__global__ __launch_bounds__(256) void gemm_bias_act(
    const unsigned short* __restrict__ A, const unsigned short* __restrict__ Bt,
    const float* __restrict__ bias, unsigned short* __restrict__ Out,
    int M, int N, int K, int relu)
{
  __shared__ alignas(16) unsigned short As[TM * TK];
  __shared__ alignas(16) unsigned short Bs[TN * TK];
  const int tileN = blockIdx.x * TN;
  const int tileM = blockIdx.y * TM;
  v8f acc[2][4] = {};
  wmma_mainloop(A, Bt, M, K, tileM, tileN, As, Bs, acc);

  const int lane  = threadIdx.x & 31;
  const int wave  = threadIdx.x >> 5;
  const int l15   = lane & 15;
  const int halfl = lane >> 4;
  const int mw    = wave >> 1;
  const int nw    = wave & 1;
#pragma unroll
  for (int mi = 0; mi < 2; ++mi) {
    const int mbase = tileM + mw * 32 + mi * 16 + halfl * 8;  // vgpr r -> M=r(+8)
#pragma unroll
    for (int nt = 0; nt < 4; ++nt) {
      const int col = tileN + nw * 64 + nt * 16 + l15;
      const float bv = bias[col];
#pragma unroll
      for (int r = 0; r < 8; ++r) {
        const int row = mbase + r;
        if (row < M) {
          float v = acc[mi][nt][r] + bv;
          if (relu) v = fmaxf(v, 0.0f);
          Out[(size_t)row * N + col] = f2bf(v);
        }
      }
    }
  }
}

// GEMM with fused scatter epilogue for the 512|128|512 split of the 1152-wide out:
//   col<512   : atomicAdd into pooled[s_idx[row]]
//   col<640   : new pred vecs (bf16)
//   else      : atomicAdd into pooled[o_idx[row]]
__global__ __launch_bounds__(256) void gemm_scatter(
    const unsigned short* __restrict__ A, const unsigned short* __restrict__ Bt,
    const float* __restrict__ bias, const int* __restrict__ triples,
    float* __restrict__ pooled, unsigned short* __restrict__ pred_out,
    int M, int K)
{
  __shared__ alignas(16) unsigned short As[TM * TK];
  __shared__ alignas(16) unsigned short Bs[TN * TK];
  const int tileN = blockIdx.x * TN;   // N = 1152 fixed
  const int tileM = blockIdx.y * TM;
  v8f acc[2][4] = {};
  wmma_mainloop(A, Bt, M, K, tileM, tileN, As, Bs, acc);

  const int lane  = threadIdx.x & 31;
  const int wave  = threadIdx.x >> 5;
  const int l15   = lane & 15;
  const int halfl = lane >> 4;
  const int mw    = wave >> 1;
  const int nw    = wave & 1;
#pragma unroll
  for (int mi = 0; mi < 2; ++mi) {
    const int mbase = tileM + mw * 32 + mi * 16 + halfl * 8;
#pragma unroll
    for (int nt = 0; nt < 4; ++nt) {
      const int col = tileN + nw * 64 + nt * 16 + l15;
      const float bv = bias[col];
#pragma unroll
      for (int r = 0; r < 8; ++r) {
        const int row = mbase + r;
        if (row < M) {
          float v = acc[mi][nt][r] + bv;
          if (col < 512) {
            atomicAdd(&pooled[(size_t)triples[row * 3 + 0] * 512 + col], v);
          } else if (col < 640) {
            pred_out[(size_t)row * 128 + (col - 512)] = f2bf(v);
          } else {
            atomicAdd(&pooled[(size_t)triples[row * 3 + 2] * 512 + (col - 640)], v);
          }
        }
      }
    }
  }
}

// ---------------- elementwise / utility kernels ----------------
__global__ void k_zero(float* p, long long n) {
  long long i = (long long)blockIdx.x * blockDim.x + threadIdx.x;
  long long st = (long long)gridDim.x * blockDim.x;
  for (; i < n; i += st) p[i] = 0.0f;
}
__global__ void k_counts(const int* __restrict__ triples, float* counts, int T) {
  int i = blockIdx.x * blockDim.x + threadIdx.x;
  if (i < T) {
    atomicAdd(&counts[triples[i * 3 + 0]], 1.0f);
    atomicAdd(&counts[triples[i * 3 + 2]], 1.0f);
  }
}
__global__ void k_recip(float* c, int n) {
  int i = blockIdx.x * blockDim.x + threadIdx.x;
  if (i < n) c[i] = 1.0f / fmaxf(c[i], 1.0f);
}
// generic embedding gather -> bf16 (stride/off pick the index column)
__global__ void k_embed(const int* __restrict__ idx_arr, int stride, int off,
                        const float* __restrict__ emb, unsigned short* __restrict__ out,
                        int n, int d) {
  long long tot = (long long)n * d;
  long long i = (long long)blockIdx.x * blockDim.x + threadIdx.x;
  long long st = (long long)gridDim.x * blockDim.x;
  for (; i < tot; i += st) {
    int row = (int)(i / d), c = (int)(i % d);
    int e = idx_arr[(size_t)row * stride + off];
    out[i] = f2bf(emb[(size_t)e * d + c]);
  }
}
// cur = [obj[s] | pred | obj[o]]  (bf16 copies)
__global__ void k_gather_cur(const int* __restrict__ triples,
                             const unsigned short* __restrict__ obj,
                             const unsigned short* __restrict__ pred,
                             unsigned short* __restrict__ cur, int T, int din) {
  long long tot = (long long)T * 3 * din;
  long long i = (long long)blockIdx.x * blockDim.x + threadIdx.x;
  long long st = (long long)gridDim.x * blockDim.x;
  for (; i < tot; i += st) {
    int t = (int)(i / (3 * din)), c = (int)(i % (3 * din));
    unsigned short v;
    if (c < din)           v = obj[(size_t)triples[t * 3 + 0] * din + c];
    else if (c < 2 * din)  v = pred[(size_t)t * din + (c - din)];
    else                   v = obj[(size_t)triples[t * 3 + 2] * din + (c - 2 * din)];
    cur[i] = v;
  }
}
__global__ void k_norm(const float* __restrict__ pooled, const float* __restrict__ inv,
                       unsigned short* __restrict__ out, long long n) {
  long long i = (long long)blockIdx.x * blockDim.x + threadIdx.x;
  long long st = (long long)gridDim.x * blockDim.x;
  for (; i < n; i += st) out[i] = f2bf(pooled[i] * inv[i >> 9]);  // row = i/512
}
// fp32 weight (K,N) -> bf16 transposed (N,K)
__global__ void k_wt(const float* __restrict__ w, unsigned short* __restrict__ wT,
                     int K, int N) {
  long long tot = (long long)K * N;
  long long i = (long long)blockIdx.x * blockDim.x + threadIdx.x;
  long long st = (long long)gridDim.x * blockDim.x;
  for (; i < tot; i += st) {
    int k = (int)(i / N), n = (int)(i % N);
    wT[(size_t)n * K + k] = f2bf(w[i]);
  }
}
// final 512 -> 4 projection, fp32 out
__global__ void k_boxes(const unsigned short* __restrict__ hbox,
                        const float* __restrict__ w2, const float* __restrict__ b2,
                        float* __restrict__ out, int O) {
  int row = blockIdx.x * blockDim.x + threadIdx.x;
  if (row >= O) return;
  float a0 = b2[0], a1 = b2[1], a2 = b2[2], a3 = b2[3];
  const unsigned short* h = hbox + (size_t)row * 512;
  for (int k = 0; k < 512; ++k) {
    float hv = bf2f(h[k]);
    a0 += hv * w2[k * 4 + 0];
    a1 += hv * w2[k * 4 + 1];
    a2 += hv * w2[k * 4 + 2];
    a3 += hv * w2[k * 4 + 3];
  }
  float* o = out + (size_t)row * 4;
  o[0] = a0; o[1] = a1; o[2] = a2; o[3] = a3;
}

// ---------------- host launcher ----------------
static int find_by_size(const int* s, int lo, int hi, int size, int occ) {
  int seen = 0;
  for (int i = lo; i < hi; ++i)
    if (s[i] == size) { if (seen == occ) return i; ++seen; }
  return lo;
}
static unsigned int eblocks(long long n) {
  long long b = (n + 255) / 256;
  if (b > 262144) b = 262144;            // grid-stride covers the rest
  return (unsigned int)b;
}

extern "C" void kernel_launch(void* const* d_in, const int* in_sizes, int n_in,
                              void* d_out, int out_size, void* d_ws, size_t ws_size,
                              hipStream_t stream) {
  (void)n_in; (void)out_size; (void)ws_size;
  // ---- top-level inputs (dict insertion order) ----
  const int*   objs     = (const int*)  d_in[0];
  const int*   triples  = (const int*)  d_in[1];
  const float* obj_emb  = (const float*)d_in[2];
  const float* pred_emb = (const float*)d_in[3];
  const int O = in_sizes[0];
  const int T = in_sizes[1] / 3;

  // ---- g0 leaves (indices 4..11), resolved by element count ----
  const float* g0_w1a = (const float*)d_in[find_by_size(in_sizes, 4, 12, 192 * 512, 0)];
  const float* g0_b1a = (const float*)d_in[find_by_size(in_sizes, 4, 12, 512, 0)];
  const float* g0_w1b = (const float*)d_in[find_by_size(in_sizes, 4, 12, 512 * 1152, 0)];
  const float* g0_b1b = (const float*)d_in[find_by_size(in_sizes, 4, 12, 1152, 0)];
  const float* g0_w2a = (const float*)d_in[find_by_size(in_sizes, 4, 12, 512 * 512, 0)];
  const float* g0_b2a = (const float*)d_in[find_by_size(in_sizes, 4, 12, 512, 1)];
  const float* g0_w2b = (const float*)d_in[find_by_size(in_sizes, 4, 12, 512 * 128, 0)];
  const float* g0_b2b = (const float*)d_in[find_by_size(in_sizes, 4, 12, 128, 0)];
  // ---- gn leaves (stacked x4, indices 12..19) ----
  const float* gn_w1a = (const float*)d_in[find_by_size(in_sizes, 12, 20, 4 * 384 * 512, 0)];
  const float* gn_b1a = (const float*)d_in[find_by_size(in_sizes, 12, 20, 4 * 512, 0)];
  const float* gn_w1b = (const float*)d_in[find_by_size(in_sizes, 12, 20, 4 * 512 * 1152, 0)];
  const float* gn_b1b = (const float*)d_in[find_by_size(in_sizes, 12, 20, 4 * 1152, 0)];
  const float* gn_w2a = (const float*)d_in[find_by_size(in_sizes, 12, 20, 4 * 512 * 512, 0)];
  const float* gn_b2a = (const float*)d_in[find_by_size(in_sizes, 12, 20, 4 * 512, 1)];
  const float* gn_w2b = (const float*)d_in[find_by_size(in_sizes, 12, 20, 4 * 512 * 128, 0)];
  const float* gn_b2b = (const float*)d_in[find_by_size(in_sizes, 12, 20, 4 * 128, 0)];
  // ---- box head (indices 20..23) ----
  const float* box_w1 = (const float*)d_in[find_by_size(in_sizes, 20, 24, 128 * 512, 0)];
  const float* box_b1 = (const float*)d_in[find_by_size(in_sizes, 20, 24, 512, 0)];
  const float* box_w2 = (const float*)d_in[find_by_size(in_sizes, 20, 24, 512 * 4, 0)];
  const float* box_b2 = (const float*)d_in[find_by_size(in_sizes, 20, 24, 4, 0)];

  // ---- workspace carve-up ----
  char* ws = (char*)d_ws;
  size_t off = 0;
  auto alloc = [&](size_t bytes) -> void* {
    void* p = ws + off;
    off += (bytes + 255) & ~(size_t)255;
    return p;
  };
  unsigned short* g0_w1aT = (unsigned short*)alloc((size_t)512 * 192 * 2);
  unsigned short* g0_w1bT = (unsigned short*)alloc((size_t)1152 * 512 * 2);
  unsigned short* g0_w2aT = (unsigned short*)alloc((size_t)512 * 512 * 2);
  unsigned short* g0_w2bT = (unsigned short*)alloc((size_t)128 * 512 * 2);
  unsigned short* gn_w1aT = (unsigned short*)alloc((size_t)4 * 512 * 384 * 2);
  unsigned short* gn_w1bT = (unsigned short*)alloc((size_t)4 * 1152 * 512 * 2);
  unsigned short* gn_w2aT = (unsigned short*)alloc((size_t)4 * 512 * 512 * 2);
  unsigned short* gn_w2bT = (unsigned short*)alloc((size_t)4 * 128 * 512 * 2);
  unsigned short* boxW1T  = (unsigned short*)alloc((size_t)512 * 128 * 2);
  float*          counts  = (float*)alloc((size_t)O * 4);
  unsigned short* objv[2]  = { (unsigned short*)alloc((size_t)O * 128 * 2),
                               (unsigned short*)alloc((size_t)O * 128 * 2) };
  unsigned short* predv[2] = { (unsigned short*)alloc((size_t)T * 128 * 2),
                               (unsigned short*)alloc((size_t)T * 128 * 2) };
  unsigned short* cur     = (unsigned short*)alloc((size_t)T * 384 * 2);
  unsigned short* tbuf    = (unsigned short*)alloc((size_t)T * 512 * 2);
  float*          pooled  = (float*)alloc((size_t)O * 512 * 4);
  unsigned short* pooledb = (unsigned short*)alloc((size_t)O * 512 * 2);
  unsigned short* t2      = (unsigned short*)alloc((size_t)O * 512 * 2);

  // ---- weight conversion (fp32 -> bf16, transposed to N x K) ----
  k_wt<<<eblocks((long long)192 * 512),  256, 0, stream>>>(g0_w1a, g0_w1aT, 192, 512);
  k_wt<<<eblocks((long long)512 * 1152), 256, 0, stream>>>(g0_w1b, g0_w1bT, 512, 1152);
  k_wt<<<eblocks((long long)512 * 512),  256, 0, stream>>>(g0_w2a, g0_w2aT, 512, 512);
  k_wt<<<eblocks((long long)512 * 128),  256, 0, stream>>>(g0_w2b, g0_w2bT, 512, 128);
  for (int l = 0; l < 4; ++l) {
    k_wt<<<eblocks((long long)384 * 512),  256, 0, stream>>>(gn_w1a + (size_t)l * 384 * 512, gn_w1aT + (size_t)l * 512 * 384, 384, 512);
    k_wt<<<eblocks((long long)512 * 1152), 256, 0, stream>>>(gn_w1b + (size_t)l * 512 * 1152, gn_w1bT + (size_t)l * 1152 * 512, 512, 1152);
    k_wt<<<eblocks((long long)512 * 512),  256, 0, stream>>>(gn_w2a + (size_t)l * 512 * 512, gn_w2aT + (size_t)l * 512 * 512, 512, 512);
    k_wt<<<eblocks((long long)512 * 128),  256, 0, stream>>>(gn_w2b + (size_t)l * 512 * 128, gn_w2bT + (size_t)l * 128 * 512, 512, 128);
  }
  k_wt<<<eblocks((long long)128 * 512), 256, 0, stream>>>(box_w1, boxW1T, 128, 512);

  // ---- counts (layer-invariant) ----
  k_zero<<<eblocks(O), 256, 0, stream>>>(counts, O);
  k_counts<<<(T + 255) / 256, 256, 0, stream>>>(triples, counts, T);
  k_recip<<<(O + 255) / 256, 256, 0, stream>>>(counts, O);

  // ---- embedding gathers -> bf16 ----
  k_embed<<<eblocks((long long)O * 64), 256, 0, stream>>>(objs, 1, 0, obj_emb, objv[0], O, 64);
  k_embed<<<eblocks((long long)T * 64), 256, 0, stream>>>(triples, 3, 1, pred_emb, predv[0], T, 64);

  // ---- 5 GraphTripleConv layers ----
  for (int l = 0; l < 5; ++l) {
    const int din = (l == 0) ? 64 : 128;
    const unsigned short* oin  = objv[l & 1];
    const unsigned short* pin  = predv[l & 1];
    unsigned short*       oout = objv[(l + 1) & 1];
    unsigned short*       pout = predv[(l + 1) & 1];

    const unsigned short *w1aT, *w1bT, *w2aT, *w2bT;
    const float *b1a, *b1b, *b2a, *b2b;
    if (l == 0) {
      w1aT = g0_w1aT; w1bT = g0_w1bT; w2aT = g0_w2aT; w2bT = g0_w2bT;
      b1a = g0_b1a; b1b = g0_b1b; b2a = g0_b2a; b2b = g0_b2b;
    } else {
      const int j = l - 1;
      w1aT = gn_w1aT + (size_t)j * 512 * 384;
      w1bT = gn_w1bT + (size_t)j * 1152 * 512;
      w2aT = gn_w2aT + (size_t)j * 512 * 512;
      w2bT = gn_w2bT + (size_t)j * 128 * 512;
      b1a = gn_b1a + (size_t)j * 512;
      b1b = gn_b1b + (size_t)j * 1152;
      b2a = gn_b2a + (size_t)j * 512;
      b2b = gn_b2b + (size_t)j * 128;
    }

    k_gather_cur<<<eblocks((long long)T * 3 * din), 256, 0, stream>>>(triples, oin, pin, cur, T, din);
    // t = relu(cur @ w1a + b1a)    [T x 512]
    dim3 g1(512 / TN, (T + TM - 1) / TM);
    gemm_bias_act<<<g1, 256, 0, stream>>>(cur, w1aT, b1a, tbuf, T, 512, 3 * din, 1);
    // pooled scatter of t @ w1b + b1b  [T x 1152]
    k_zero<<<eblocks((long long)O * 512), 256, 0, stream>>>(pooled, (long long)O * 512);
    dim3 g2(1152 / TN, (T + TM - 1) / TM);
    gemm_scatter<<<g2, 256, 0, stream>>>(tbuf, w1bT, b1b, triples, pooled, pout, T, 512);
    // pooled /= counts, -> bf16
    k_norm<<<eblocks((long long)O * 512), 256, 0, stream>>>(pooled, counts, pooledb, (long long)O * 512);
    // t2 = relu(pooled @ w2a + b2a)   [O x 512]
    dim3 g3(512 / TN, (O + TM - 1) / TM);
    gemm_bias_act<<<g3, 256, 0, stream>>>(pooledb, w2aT, b2a, t2, O, 512, 512, 1);
    // new_obj = t2 @ w2b + b2b        [O x 128]
    dim3 g4(128 / TN, (O + TM - 1) / TM);
    gemm_bias_act<<<g4, 256, 0, stream>>>(t2, w2bT, b2b, oout, O, 128, 512, 0);
  }

  // ---- box head ----
  const unsigned short* ofin = objv[1];   // (5 & 1) == 1 after 5 layers
  dim3 gb(512 / TN, (O + TM - 1) / TM);
  gemm_bias_act<<<gb, 256, 0, stream>>>(ofin, boxW1T, box_b1, t2, O, 512, 128, 1);
  k_boxes<<<(O + 255) / 256, 256, 0, stream>>>(t2, box_w2, box_b2, (float*)d_out, O);
}
